// NeuralCTLSTM_27230092657708
// MI455X (gfx1250) — matrix operand order repaired and checked by reading, not compiled, not run on hardware
//
#include <hip/hip_runtime.h>
#include <hip/hip_bf16.h>

typedef __attribute__((ext_vector_type(4)))  float    v4f;
typedef __attribute__((ext_vector_type(8)))  float    v8f;
typedef __attribute__((ext_vector_type(4)))  _Float16 v4h;
typedef __attribute__((ext_vector_type(8)))  _Float16 v8h;
typedef __attribute__((ext_vector_type(16))) _Float16 v16h;

#define B_SZ   65536
#define H_SZ   256
#define G_SZ   7
#define NT     16          // output columns per block
#define WSTRH  264         // padded LDS row stride in halfs (528B: 16B-aligned, bank-spread)
#define BH     ((size_t)B_SZ * H_SZ)

__device__ __forceinline__ float fast_sigmoid(float x) {
    return 1.0f / (1.0f + __expf(-x));
}
__device__ __forceinline__ float fast_tanh(float x) {
    x = fminf(fmaxf(x, -15.0f), 15.0f);
    float e = __expf(2.0f * x);
    return (e - 1.0f) / (e + 1.0f);
}
__device__ __forceinline__ float fast_softplus(float x) {
    return fmaxf(x, 0.0f) + __logf(1.0f + __expf(-fabsf(x)));
}
__device__ __forceinline__ v16h combine16(v8h lo, v8h hi) {
    v16h r;
    #pragma unroll
    for (int i = 0; i < 8; ++i) { r[i] = lo[i]; r[8 + i] = hi[i]; }
    return r;
}

// ---- Pass 1: f32 -> f16 conversion into workspace (vectorized, grid-stride-free) ----
__global__ void cvt_f32_f16(const float* __restrict__ src, _Float16* __restrict__ dst, int n4) {
    int i = blockIdx.x * blockDim.x + threadIdx.x;
    if (i < n4) {
        v4f v = *(const v4f*)(src + (size_t)i * 4);
        v4h o;
        o[0] = (_Float16)v[0]; o[1] = (_Float16)v[1];
        o[2] = (_Float16)v[2]; o[3] = (_Float16)v[3];
        *(v4h*)(dst + (size_t)i * 4) = o;
    }
}

// ---- Pass 2: fused 7-gate f16 WMMA GEMM + CT-LSTM epilogue ----
__launch_bounds__(256)
__global__ void ctlstm_fused_wmma_f16(const float* __restrict__ dt,
                                      const _Float16* __restrict__ hf,
                                      const float* __restrict__ c_ti,
                                      const float* __restrict__ cbar,
                                      const _Float16* __restrict__ Wf,
                                      const float* __restrict__ bias,
                                      float* __restrict__ out)
{
    // W slices for this 16-column tile, all 7 gates: [g][n][k] f16, padded k-stride
    __shared__ _Float16 lds_w[G_SZ * NT * WSTRH];

    const int tid   = threadIdx.x;
    const int wave  = tid >> 5;
    const int lane  = tid & 31;
    const int l15   = lane & 15;
    const int hsel  = lane >> 4;                  // 0 or 1: which half-wave

    const int n0 = blockIdx.x * NT;               // output-column tile base
    const int m0 = (blockIdx.y * 8 + wave) * 16;  // this wave's batch tile base

    // ---- Stage Wf[g][n0+r][0..255] into LDS (16B chunks, coalesced) ----
    for (int t = tid; t < G_SZ * NT * (H_SZ / 8); t += 256) {
        const int g = t / (NT * (H_SZ / 8));
        const int r = (t / (H_SZ / 8)) % NT;
        const int c = t % (H_SZ / 8);
        v8h v = *(const v8h*)(Wf + (size_t)(g * H_SZ + n0 + r) * H_SZ + c * 8);
        *(v8h*)(&lds_w[(g * NT + r) * WSTRH + c * 8]) = v;
    }
    __syncthreads();

    // ---- K-loop: 8 steps of V_WMMA_F32_16X16X32_F16, 7 gate accumulators ----
    v8f acc[G_SZ];
    #pragma unroll
    for (int g = 0; g < G_SZ; ++g) acc[g] = (v8f){0,0,0,0,0,0,0,0};

    // A frag (16-bit 16x32 layout): lane holds K = 8*hsel..+7 and 16+8*hsel..+7
    const _Float16* aptr = hf + (size_t)(m0 + l15) * H_SZ + 8 * hsel;

    #pragma unroll
    for (int k0 = 0; k0 < H_SZ; k0 += 32) {
        v8h alo = *(const v8h*)(aptr + k0);
        v8h ahi = *(const v8h*)(aptr + k0 + 16);
        v16h a = combine16(alo, ahi);
        #pragma unroll
        for (int g = 0; g < G_SZ; ++g) {
            // B frag (f16 32x16): lane = col N=l15, K = 8*hsel..+7 and 16+8*hsel..+7
            const _Float16* bp = &lds_w[(g * NT + l15) * WSTRH + k0 + 8 * hsel];
            v8h blo = *(const v8h*)(bp);
            v8h bhi = *(const v8h*)(bp + 16);
            v16h b = combine16(blo, bhi);
            acc[g] = __builtin_amdgcn_wmma_f32_16x16x32_f16(
                false, a, false, b, (short)0, acc[g], false, false);
        }
    }

    // ---- Fused epilogue ----
    // C/D layout: element i of v8f -> M = i + 8*hsel, N = l15
    const int ocol  = n0 + l15;
    const int mbase = m0 + (hsel << 3);

    float bg[G_SZ];
    #pragma unroll
    for (int g = 0; g < G_SZ; ++g) bg[g] = bias[g * H_SZ + ocol];

    #pragma unroll
    for (int i = 0; i < 8; ++i) {
        const int brow = mbase + i;
        const size_t idx = (size_t)brow * H_SZ + ocol;

        const float ig    = fast_sigmoid(acc[0][i] + bg[0]);
        const float fg    = fast_sigmoid(acc[1][i] + bg[1]);
        const float og    = fast_sigmoid(acc[2][i] + bg[2]);
        const float ibarg = fast_sigmoid(acc[3][i] + bg[3]);
        const float fbarg = fast_sigmoid(acc[4][i] + bg[4]);
        const float z     = fast_tanh   (acc[5][i] + bg[5]);
        const float decay = fast_softplus(acc[6][i] + bg[6]);

        const float d  = dt[brow];
        const float c0 = c_ti[idx];
        const float cb = cbar[idx];

        const float c_after = cb + (c0 - cb) * __expf(-decay * d);
        const float c_new   = fg * c_after + ig * z;
        const float cbarn   = fbarg * cb + ibarg * z;
        const float h_new   = og * fast_tanh(c_after);

        out[0 * BH + idx] = og;
        out[1 * BH + idx] = h_new;
        out[2 * BH + idx] = c_new;
        out[3 * BH + idx] = cbarn;
        out[4 * BH + idx] = decay;
    }
}

extern "C" void kernel_launch(void* const* d_in, const int* in_sizes, int n_in,
                              void* d_out, int out_size, void* d_ws, size_t ws_size,
                              hipStream_t stream) {
    const float* dt    = (const float*)d_in[0];  // inter_times (B)
    const float* h_ti  = (const float*)d_in[1];  // (B,H)
    const float* c_ti  = (const float*)d_in[2];  // (B,H)
    const float* cbar  = (const float*)d_in[3];  // (B,H)
    const float* W     = (const float*)d_in[4];  // (G,H,H)
    const float* bias  = (const float*)d_in[5];  // (G,H)
    float* out = (float*)d_out;                  // 5 x (B,H) concatenated

    _Float16* hf = (_Float16*)d_ws;              // 32 MB
    _Float16* Wf = hf + BH;                      // 0.875 MB

    // Pass 1: downconvert h and W to f16 in workspace
    {
        int n4h = (int)(BH / 4);                     // 4,194,304
        int n4w = (G_SZ * H_SZ * H_SZ) / 4;          // 114,688
        cvt_f32_f16<<<(n4h + 255) / 256, 256, 0, stream>>>(h_ti, hf, n4h);
        cvt_f32_f16<<<(n4w + 255) / 256, 256, 0, stream>>>(W, Wf, n4w);
    }

    // Pass 2: fused WMMA GEMM + epilogue
    dim3 grid(H_SZ / NT, B_SZ / (16 * 8));       // 16 column tiles x 512 batch groups
    dim3 block(256);
    ctlstm_fused_wmma_f16<<<grid, block, 0, stream>>>(dt, hf, c_ti, cbar, Wf, bias, out);
}
